// TransformerLatentODEWrapper_55594056680194
// MI455X (gfx1250) — matrix operand order repaired and checked by reading, not compiled, not run on hardware
//
#include <hip/hip_runtime.h>
#include <hip/hip_bf16.h>

typedef _Float16 h16;
typedef __attribute__((ext_vector_type(16))) _Float16 v16h;
typedef __attribute__((ext_vector_type(8)))  _Float16 v8h;
typedef __attribute__((ext_vector_type(4)))  _Float16 v4h;
typedef __attribute__((ext_vector_type(8)))  float    v8f;
typedef __attribute__((ext_vector_type(4)))  unsigned v4u;
typedef __attribute__((ext_vector_type(8)))  int      v8i;

#define DEV __device__ __forceinline__

DEV v8f zero8() { v8f z = {0.f,0.f,0.f,0.f,0.f,0.f,0.f,0.f}; return z; }

DEV v8f wmma32(v16h a, v16h b, v8f c) {
    // D = A(16x32 f16) * B(32x16 f16) + C(16x16 f32)
    return __builtin_amdgcn_wmma_f32_16x16x32_f16(false, a, false, b, (short)0, c, false, false);
}

// A-fragment (16x32, row-major source): lane holds row = (m0 + lane%16),
// halves K = k0 + {8g..8g+7, 16+8g..16+8g+7}, g = lane/16.
DEV v16h load_afrag(const h16* base, int ld, int row, int k0, int g) {
    const h16* p = base + (size_t)row * ld + k0 + 8 * g;
    v8h lo = *(const v8h*)p;
    v8h hi = *(const v8h*)(p + 16);
    return __builtin_shufflevector(lo, hi, 0,1,2,3,4,5,6,7,8,9,10,11,12,13,14,15);
}

// B-fragment (32x16) from a row-major (N x K) "weight" matrix:
// lane holds col n = n0 + lane%16, halves K = k0 + {8g..8g+7, 16+8g..16+8g+7}.
DEV v16h load_bfrag(const h16* base, int ld, int n, int k0, int g) {
    const h16* p = base + (size_t)n * ld + k0 + 8 * g;
    v8h lo = *(const v8h*)p;
    v8h hi = *(const v8h*)(p + 16);
    return __builtin_shufflevector(lo, hi, 0,1,2,3,4,5,6,7,8,9,10,11,12,13,14,15);
}

// ---------------------------------------------------------------------------
// Tensor Data Mover: async 2D tile load Global -> LDS (D# per ISA ch.8).
// Loads `rows` x `row_elems` f16 tile with source row stride `stride_elems`,
// padding 4 DWORDs (8 halves) after each 64-half row -> LDS stride 72 halves.
// Issued once per calling wave; tracked by TENSORcnt.
// ---------------------------------------------------------------------------
DEV void tdm_load_2d(unsigned lds_addr, const void* gaddr,
                     unsigned rows, unsigned row_elems, unsigned long long stride_elems)
{
    unsigned long long ga = (unsigned long long)gaddr;
    v4u g0;
    g0.x = 1u;                                         // count=1, user descriptor
    g0.y = lds_addr;                                   // LDS byte address
    g0.z = (unsigned)(ga & 0xFFFFFFFFu);               // global_addr[31:0]
    g0.w = (unsigned)((ga >> 32) & 0x1FFFFFFu)         // global_addr[56:32]
         | (2u << 30);                                 // type = 2 ("image")
    const unsigned td0 = 1u << 20, td1 = 1u << 20;     // huge tensor dims: no clip
    v8i g1;
    g1[0] = (int)((1u << 16)                           // data_size = 2 bytes
                | (1u << 20)                           // pad_enable
                | (4u << 22)                           // pad_interval: 32 DWORDs (one row)
                | (3u << 25));                         // pad_amount: 4 DWORDs
    g1[1] = (int)((td0 & 0xFFFFu) << 16);              // [47:32]=0, tensor_dim0 lo
    g1[2] = (int)(((td0 >> 16) & 0xFFFFu) | ((td1 & 0xFFFFu) << 16));
    g1[3] = (int)(((td1 >> 16) & 0xFFFFu) | (row_elems << 16));   // tile_dim0
    g1[4] = (int)rows;                                 // tile_dim1 (tile_dim2 = 0)
    g1[5] = (int)(unsigned)(stride_elems & 0xFFFFFFFFull);        // dim0_stride lo
    g1[6] = (int)(unsigned)((stride_elems >> 32) & 0xFFFFull);    // dim0_stride hi
    g1[7] = 0;                                         // dim1_stride = 0 (2D)
    asm volatile("tensor_load_to_lds %0, %1" :: "s"(g0), "s"(g1) : "memory");
}

// ---------------------------------------------------------------------------
// WMMA GEMM with TDM-staged, double-buffered LDS tiles.
// C[M,N] = A[M,K] @ W[N,K]^T + bias (+relu), f32 or f16 out.
// Block: 256 thr = 8 waves (2 m-tiles x 4 n-tiles) -> 32x64 block tile,
// K-slab = 64. Requires M%32==0, N%64==0, K%64==0 (true at all call sites).
// ---------------------------------------------------------------------------
template<int STOREH, int RELU>
__global__ __launch_bounds__(256) void gemm_tdm(
    const h16* __restrict__ A, int lda,
    const h16* __restrict__ W, int ldb, int K,
    const float* __restrict__ bias,
    float* __restrict__ Cf, h16* __restrict__ Ch, int ldc)
{
    __shared__ h16 At[2][32 * 72];
    __shared__ h16 Bt[2][64 * 72];
    const int lane = threadIdx.x & 31;
    const int wid  = threadIdx.x >> 5;
    const int g    = lane >> 4;
    const int m0   = blockIdx.y * 32;
    const int n0   = blockIdx.x * 64;
    const int ml   = (wid >> 2) * 16 + (lane & 15);   // row in A tile
    const int nl   = (wid & 3) * 16 + (lane & 15);    // row in B tile
    const int col  = n0 + (wid & 3) * 16 + (lane & 15);

    if (wid == 0) {
        tdm_load_2d((unsigned)(size_t)&At[0][0], A + (size_t)m0 * lda, 32, 64, (unsigned long long)lda);
        tdm_load_2d((unsigned)(size_t)&Bt[0][0], W + (size_t)n0 * ldb, 64, 64, (unsigned long long)ldb);
    }
    v8f acc = zero8();
    int cur = 0;
    for (int k0 = 0; k0 < K; k0 += 64) {
        if (wid == 0) __builtin_amdgcn_s_wait_tensorcnt(0);
        __syncthreads();                                // buf[cur] visible to all
        if (wid == 0 && (k0 + 64) < K) {                // prefetch next K-slab
            tdm_load_2d((unsigned)(size_t)&At[cur ^ 1][0],
                        A + (size_t)m0 * lda + k0 + 64, 32, 64, (unsigned long long)lda);
            tdm_load_2d((unsigned)(size_t)&Bt[cur ^ 1][0],
                        W + (size_t)n0 * ldb + k0 + 64, 64, 64, (unsigned long long)ldb);
        }
        const h16* at = At[cur];
        const h16* bt = Bt[cur];
        acc = wmma32(load_afrag(at, 72, ml, 0, g),  load_bfrag(bt, 72, nl, 0, g),  acc);
        acc = wmma32(load_afrag(at, 72, ml, 32, g), load_bfrag(bt, 72, nl, 32, g), acc);
        __syncthreads();                                // done reading buf[cur]
        cur ^= 1;
    }
    const float bn = bias ? bias[col] : 0.f;
    const int mbase = m0 + (wid >> 2) * 16;
#pragma unroll
    for (int r = 0; r < 8; ++r) {
        float v = acc[r] + bn;
        if (RELU) v = fmaxf(v, 0.f);
        size_t idx = (size_t)(mbase + r + 8 * g) * ldc + col;
        if (STOREH) Ch[idx] = (h16)v;
        else        Cf[idx] = v;
    }
}

// ---------------------------------------------------------------------------
// f32 -> f16 conversion
// ---------------------------------------------------------------------------
__global__ void cvt_f32_f16(const float* __restrict__ in, h16* __restrict__ out, int n)
{
    int i = blockIdx.x * 256 + threadIdx.x;
    if (i < n) out[i] = (h16)in[i];
}

// ---------------------------------------------------------------------------
// Sinusoidal PE add: xf += pe(t,n); also write f16 copy. M = 8192, D = 1024.
// pe[t, i]      = sin(t * 10000^(-2i/1024))   i < 512
// pe[t, 512+i]  = cos(t * 10000^(-2i/1024))
// ---------------------------------------------------------------------------
__global__ void pe_add(float* __restrict__ xf, h16* __restrict__ xh)
{
    int idx = blockIdx.x * 256 + threadIdx.x;     // exactly 8192*1024 threads
    int n = idx & 1023;
    int t = (idx >> 10) & 255;
    int i2 = (n < 512) ? n : (n - 512);
    float freq = expf(-9.210340371976184f * ((float)(2 * i2) * (1.f / 1024.f)));
    float ang  = (float)t * freq;
    float pe   = (n < 512) ? sinf(ang) : cosf(ang);
    float v = xf[idx] + pe;
    xf[idx] = v;
    xh[idx] = (h16)v;
}

// ---------------------------------------------------------------------------
// Residual + LayerNorm over D=1024. One block per row.
// ---------------------------------------------------------------------------
__global__ __launch_bounds__(256) void resid_ln(
    const float* xin, const float* delta,
    const float* __restrict__ s, const float* __restrict__ bb,
    float* xf, h16* __restrict__ xh)
{
    const int m = blockIdx.x;
    const int tid = threadIdx.x;
    const size_t off = (size_t)m * 1024;
    float vals[4];
    float sum = 0.f, sq = 0.f;
#pragma unroll
    for (int i = 0; i < 4; ++i) {
        int n = tid + i * 256;
        float v = xin[off + n] + delta[off + n];
        vals[i] = v; sum += v; sq += v * v;
    }
    for (int o = 16; o; o >>= 1) {
        sum += __shfl_xor(sum, o, 32);
        sq  += __shfl_xor(sq,  o, 32);
    }
    __shared__ float rs[8], rq[8];
    __shared__ float stats[2];
    if ((tid & 31) == 0) { rs[tid >> 5] = sum; rq[tid >> 5] = sq; }
    __syncthreads();
    if (tid == 0) {
        float S = 0.f, Q = 0.f;
        for (int w = 0; w < 8; ++w) { S += rs[w]; Q += rq[w]; }
        float mean = S * (1.f / 1024.f);
        stats[0] = mean;
        stats[1] = Q * (1.f / 1024.f) - mean * mean;
    }
    __syncthreads();
    const float mean = stats[0];
    const float inv  = rsqrtf(stats[1] + 1e-5f);
#pragma unroll
    for (int i = 0; i < 4; ++i) {
        int n = tid + i * 256;
        float y = (vals[i] - mean) * inv * s[n] + bb[n];
        xf[off + n] = y;
        xh[off + n] = (h16)y;
    }
}

// ---------------------------------------------------------------------------
// Attention per (b,h): S = Q K^T / 8, softmax, O = P V.  T=256, dh=64.
// Block = 64 threads (2 waves); each wave owns 8 query tiles of 16 rows.
// V^T staged in LDS; P staged per wave to convert C-layout -> A-layout.
// ---------------------------------------------------------------------------
__global__ __launch_bounds__(64) void attn_kernel(
    const h16* __restrict__ qkv, h16* __restrict__ obuf)
{
    __shared__ h16 VT[64 * 264];       // V transposed, padded stride 264
    __shared__ h16 PS[2][16 * 264];    // per-wave probability tile
    const int b = blockIdx.x >> 4;
    const int h = blockIdx.x & 15;
    const int tid = threadIdx.x, lane = tid & 31, wid = tid >> 5;
    const int g = lane >> 4;

    const h16* base = qkv + (size_t)b * 256 * 3072;
    // stage V^T : VT[d][t] = V[t][d], vectorized 4-half global reads
    for (int id = tid; id < 256 * 16; id += 64) {
        int t = id >> 4, dq = (id & 15) * 4;
        v4h v = *(const v4h*)(base + (size_t)t * 3072 + 2048 + h * 64 + dq);
#pragma unroll
        for (int q = 0; q < 4; ++q) VT[(dq + q) * 264 + t] = v[q];
    }
    __syncthreads();

    const h16* qb = base + h * 64;
    const h16* kb = base + 1024 + h * 64;
    h16* Pw = PS[wid];

    for (int qt = wid; qt < 16; qt += 2) {
        const int m0 = qt * 16;
        const int qrow = m0 + (lane & 15);
        v16h aq0 = load_afrag(qb, 3072, qrow, 0, g);
        v16h aq1 = load_afrag(qb, 3072, qrow, 32, g);

        v8f sc[16];
#pragma unroll
        for (int nt = 0; nt < 16; ++nt) {
            int kr = nt * 16 + (lane & 15);
            v8f c = zero8();
            c = wmma32(aq0, load_bfrag(kb, 3072, kr, 0, g), c);
            c = wmma32(aq1, load_bfrag(kb, 3072, kr, 32, g), c);
            sc[nt] = c;
        }
        // softmax(s/8) per row; row m0+r+8g lives in lanes of the same half-wave
#pragma unroll
        for (int r = 0; r < 8; ++r) {
            float mx = -3.0e38f;
#pragma unroll
            for (int nt = 0; nt < 16; ++nt) mx = fmaxf(mx, sc[nt][r]);
            for (int o = 8; o; o >>= 1) mx = fmaxf(mx, __shfl_xor(mx, o, 32));
            float sum = 0.f;
#pragma unroll
            for (int nt = 0; nt < 16; ++nt) {
                float e = expf((sc[nt][r] - mx) * 0.125f);
                sc[nt][r] = e; sum += e;
            }
            for (int o = 8; o; o >>= 1) sum += __shfl_xor(sum, o, 32);
            float inv = 1.f / sum;
#pragma unroll
            for (int nt = 0; nt < 16; ++nt) sc[nt][r] *= inv;
        }
        // C-layout -> row-major P tile in LDS
#pragma unroll
        for (int nt = 0; nt < 16; ++nt)
#pragma unroll
            for (int r = 0; r < 8; ++r)
                Pw[(r + 8 * g) * 264 + nt * 16 + (lane & 15)] = (h16)sc[nt][r];
        __builtin_amdgcn_wave_barrier();  // same-wave LDS ops are in-order

        // O = P @ V : A from P (LDS), B from VT (LDS)
        v8f oa[4] = { zero8(), zero8(), zero8(), zero8() };
        for (int k0 = 0; k0 < 256; k0 += 32) {
            v16h ap = load_afrag(Pw, 264, lane & 15, k0, g);
#pragma unroll
            for (int nt = 0; nt < 4; ++nt)
                oa[nt] = wmma32(ap, load_bfrag(VT, 264, nt * 16 + (lane & 15), k0, g), oa[nt]);
        }
        h16* orow = obuf + (size_t)(b * 256 + m0) * 1024 + h * 64;
#pragma unroll
        for (int nt = 0; nt < 4; ++nt)
#pragma unroll
            for (int r = 0; r < 8; ++r)
                orow[(size_t)(r + 8 * g) * 1024 + nt * 16 + (lane & 15)] = (h16)oa[nt][r];
    }
}

// ---------------------------------------------------------------------------
// Latent sampling + small copies
// ---------------------------------------------------------------------------
__global__ void sampler(const float* __restrict__ zp, const float* __restrict__ eps,
                        float* __restrict__ omean, float* __restrict__ ologv,
                        float* __restrict__ predz, h16* __restrict__ predzh,
                        float* __restrict__ z0f)
{
    int i = blockIdx.x * 256 + threadIdx.x;   // 2048
    if (i >= 2048) return;
    int b = i >> 6, j = i & 63;
    float mean = zp[b * 128 + j];
    float logv = zp[b * 128 + 64 + j];
    float z0 = mean + eps[i] * expf(0.5f * logv);
    omean[i] = mean;
    ologv[i] = logv;
    z0f[i] = z0;
    size_t o = (size_t)(b * 65) * 64 + j;
    predz[o] = z0;
    predzh[o] = (h16)z0;
}

__global__ void times_copy(const float* __restrict__ ft, float* __restrict__ out)
{
    int i = blockIdx.x * 256 + threadIdx.x;   // 2080
    if (i >= 2080) return;
    int b = i / 65, s = i % 65;
    out[i] = ft[b * 320 + 255 + s];
}

__global__ void extract_h(const float* __restrict__ xf, h16* __restrict__ hh)
{
    int i = blockIdx.x * 256 + threadIdx.x;   // 32768
    if (i >= 32768) return;
    int b = i >> 10, c = i & 1023;
    hh[i] = (h16)xf[(size_t)(b * 256 + 255) * 1024 + c];
}

// ---------------------------------------------------------------------------
// Latent ODE: persistent single workgroup, 8 waves. RK4, 64 intervals x 4 sub.
// f(z): 64->256 relu, 2x(256->256 relu), 256->64. Activations in LDS, fp16
// weights streamed from L2. Each 32x256 layer = 32 WMMA tiles over 8 waves.
// ---------------------------------------------------------------------------
__global__ __launch_bounds__(256) void ode_rk4(
    const h16* __restrict__ W1, const float* __restrict__ B1,
    const h16* __restrict__ Wm, const float* __restrict__ Bm,
    const h16* __restrict__ Wf, const float* __restrict__ Bf,
    const float* __restrict__ z0f, const float* __restrict__ ftime,
    float* __restrict__ predz, h16* __restrict__ predzh)
{
    __shared__ float zS[2048], accS[2048], kS[2048];
    __shared__ h16 zin[32 * 72];
    __shared__ h16 uA[32 * 264], uB[32 * 264];
    const int tid = threadIdx.x, lane = tid & 31, wid = tid >> 5;
    const int g = lane >> 4;

    for (int i = tid; i < 2048; i += 256) zS[i] = z0f[i];
    __syncthreads();

    for (int iv = 0; iv < 64; ++iv) {
        const float dt = (ftime[256 + iv] - ftime[255 + iv]) * 0.25f;
        for (int sub = 0; sub < 4; ++sub) {
            for (int i = tid; i < 2048; i += 256)
                zin[(i >> 6) * 72 + (i & 63)] = (h16)zS[i];
            __syncthreads();

            for (int stage = 0; stage < 4; ++stage) {
                // layer 1: (32x64) @ (256x64)^T
                for (int tile = wid; tile < 32; tile += 8) {
                    int mt = tile >> 4, nt = tile & 15;
                    int row = mt * 16 + (lane & 15), n = nt * 16 + (lane & 15);
                    v8f c = zero8();
                    for (int k0 = 0; k0 < 64; k0 += 32)
                        c = wmma32(load_afrag(zin, 72, row, k0, g),
                                   load_bfrag(W1, 64, n, k0, g), c);
                    float bb = B1[n];
#pragma unroll
                    for (int r = 0; r < 8; ++r)
                        uA[(mt * 16 + r + 8 * g) * 264 + n] = (h16)fmaxf(c[r] + bb, 0.f);
                }
                __syncthreads();
                // 2 mid layers: (32x256) @ (256x256)^T
                for (int j = 0; j < 2; ++j) {
                    const h16* src = j ? uB : uA;
                    h16* dst = j ? uA : uB;
                    const h16* w = Wm + j * 65536;
                    const float* bi = Bm + j * 256;
                    for (int tile = wid; tile < 32; tile += 8) {
                        int mt = tile >> 4, nt = tile & 15;
                        int row = mt * 16 + (lane & 15), n = nt * 16 + (lane & 15);
                        v8f c = zero8();
                        for (int k0 = 0; k0 < 256; k0 += 32)
                            c = wmma32(load_afrag(src, 264, row, k0, g),
                                       load_bfrag(w, 256, n, k0, g), c);
                        float bb = bi[n];
#pragma unroll
                        for (int r = 0; r < 8; ++r)
                            dst[(mt * 16 + r + 8 * g) * 264 + n] = (h16)fmaxf(c[r] + bb, 0.f);
                    }
                    __syncthreads();
                }
                // final: (32x256) @ (64x256)^T -> kS (8 tiles, 1 per wave)
                {
                    int mt = wid >> 2, nt = wid & 3;
                    int row = mt * 16 + (lane & 15), n = nt * 16 + (lane & 15);
                    v8f c = zero8();
                    for (int k0 = 0; k0 < 256; k0 += 32)
                        c = wmma32(load_afrag(uA, 264, row, k0, g),
                                   load_bfrag(Wf, 256, n, k0, g), c);
                    float bb = Bf[n];
#pragma unroll
                    for (int r = 0; r < 8; ++r)
                        kS[(mt * 16 + r + 8 * g) * 64 + n] = c[r] + bb;
                }
                __syncthreads();
                // RK4 combine
                for (int i = tid; i < 2048; i += 256) {
                    float kv = kS[i];
                    float a;
                    if (stage == 0)      a = kv;
                    else if (stage == 3) a = accS[i] + kv;
                    else                 a = accS[i] + 2.f * kv;
                    accS[i] = a;
                    if (stage < 3) {
                        float cc = (stage == 2) ? dt : 0.5f * dt;
                        zin[(i >> 6) * 72 + (i & 63)] = (h16)(zS[i] + cc * kv);
                    } else {
                        zS[i] += (dt * (1.f / 6.f)) * a;
                    }
                }
                __syncthreads();
            }
        }
        for (int i = tid; i < 2048; i += 256) {
            int b = i >> 6, j = i & 63;
            float v = zS[i];
            size_t o = (size_t)(b * 65 + 1 + iv) * 64 + j;
            predz[o] = v;
            predzh[o] = (h16)v;
        }
        __syncthreads();
    }
}

// ---------------------------------------------------------------------------
// Host orchestration
// ---------------------------------------------------------------------------
extern "C" void kernel_launch(void* const* d_in, const int* in_sizes, int n_in,
                              void* d_out, int out_size, void* d_ws, size_t ws_size,
                              hipStream_t stream)
{
    (void)in_sizes; (void)n_in; (void)out_size; (void)ws_size;
    const float* obs   = (const float*)d_in[0];
    const float* ftime = (const float*)d_in[2];
    const float* eps   = (const float*)d_in[3];
    const float* W_val = (const float*)d_in[5];
    const float* b_val = (const float*)d_in[6];
    const float* Wqkv  = (const float*)d_in[7];
    const float* bqkv  = (const float*)d_in[8];
    const float* Wo    = (const float*)d_in[9];
    const float* bo    = (const float*)d_in[10];
    const float* ln1s  = (const float*)d_in[11];
    const float* ln1b  = (const float*)d_in[12];
    const float* Wff1  = (const float*)d_in[13];
    const float* bff1  = (const float*)d_in[14];
    const float* Wff2  = (const float*)d_in[15];
    const float* bff2  = (const float*)d_in[16];
    const float* ln2s  = (const float*)d_in[17];
    const float* ln2b  = (const float*)d_in[18];
    const float* Wproj = (const float*)d_in[19];
    const float* bproj = (const float*)d_in[20];
    const float* oW1   = (const float*)d_in[21];
    const float* ob1   = (const float*)d_in[22];
    const float* oWm   = (const float*)d_in[23];
    const float* obm   = (const float*)d_in[24];
    const float* oWf   = (const float*)d_in[25];
    const float* obf   = (const float*)d_in[26];
    const float* dW1   = (const float*)d_in[27];
    const float* db1   = (const float*)d_in[28];
    const float* dWm   = (const float*)d_in[29];
    const float* dbm   = (const float*)d_in[30];
    const float* dWout = (const float*)d_in[31];
    const float* dbout = (const float*)d_in[32];

    float* out = (float*)d_out;
    float* out_predx = out;                 // (32,65,64) = 133120
    float* out_predz = out + 133120;        // (32,65,64)
    float* out_times = out + 266240;        // (32,65)
    float* out_mean  = out + 268320;        // (32,64)
    float* out_logv  = out + 270368;        // (32,64)

    char* ws = (char*)d_ws;
    size_t off = 0;
    auto take = [&](size_t bytes) -> char* {
        char* p = ws + off;
        off += (bytes + 255) & ~(size_t)255;
        return p;
    };
    h16*   wqkv_h  = (h16*)take(3072ull * 1024 * 2);
    h16*   wo_h    = (h16*)take(1024ull * 1024 * 2);
    h16*   wff1_h  = (h16*)take(4096ull * 1024 * 2);
    h16*   wff2_h  = (h16*)take(4096ull * 1024 * 2);
    h16*   xh      = (h16*)take(8192ull * 1024 * 2);
    float* xf      = (float*)take(8192ull * 1024 * 4);
    float* tmp     = (float*)take(8192ull * 1024 * 4);
    h16*   slab    = (h16*)take(8192ull * 4096 * 2);   // qkv(ld3072) then ffh(ld4096)
    h16*   obufp   = (h16*)take(8192ull * 1024 * 2);
    h16*   obs_h   = (h16*)take(8192ull * 64 * 2);
    h16*   wval_h  = (h16*)take(1024ull * 64 * 2);
    h16*   wproj_h = (h16*)take(128ull * 1024 * 2);
    h16*   h_h     = (h16*)take(32ull * 1024 * 2);
    float* zp      = (float*)take(32ull * 128 * 4);
    h16*   ow1h    = (h16*)take(256ull * 64 * 2);
    h16*   owmh    = (h16*)take(2ull * 256 * 256 * 2);
    h16*   owfh    = (h16*)take(64ull * 256 * 2);
    h16*   dw1h    = (h16*)take(512ull * 64 * 2);
    h16*   dwmh    = (h16*)take(2ull * 512 * 512 * 2);
    h16*   dwouth  = (h16*)take(64ull * 512 * 2);
    float* z0f     = (float*)take(2048ull * 4);
    h16*   predzh  = (h16*)take(2080ull * 64 * 2);
    h16*   du0     = (h16*)take(2080ull * 512 * 2);
    h16*   du1     = (h16*)take(2080ull * 512 * 2);

    auto cvt = [&](const float* src, h16* dst, int n) {
        cvt_f32_f16<<<(n + 255) / 256, 256, 0, stream>>>(src, dst, n);
    };
    cvt(obs, obs_h, 8192 * 64);
    cvt(W_val, wval_h, 1024 * 64);
    cvt(Wproj, wproj_h, 128 * 1024);
    cvt(oW1, ow1h, 256 * 64);
    cvt(oWm, owmh, 2 * 256 * 256);
    cvt(oWf, owfh, 64 * 256);
    cvt(dW1, dw1h, 512 * 64);
    cvt(dWm, dwmh, 2 * 512 * 512);
    cvt(dWout, dwouth, 64 * 512);

    // embed: xf = obs @ W_val^T + b_val ; then add PE, make f16 copy
    gemm_tdm<0,0><<<dim3(1024 / 64, 8192 / 32), 256, 0, stream>>>(
        obs_h, 64, wval_h, 64, 64, b_val, xf, nullptr, 1024);
    pe_add<<<32768, 256, 0, stream>>>(xf, xh);

    for (int l = 0; l < 6; ++l) {
        cvt(Wqkv + (size_t)l * 3072 * 1024, wqkv_h, 3072 * 1024);
        cvt(Wo   + (size_t)l * 1024 * 1024, wo_h,   1024 * 1024);
        cvt(Wff1 + (size_t)l * 4096 * 1024, wff1_h, 4096 * 1024);
        cvt(Wff2 + (size_t)l * 4096 * 1024, wff2_h, 4096 * 1024);

        gemm_tdm<1,0><<<dim3(3072 / 64, 8192 / 32), 256, 0, stream>>>(
            xh, 1024, wqkv_h, 1024, 1024, bqkv + l * 3072, nullptr, slab, 3072);
        attn_kernel<<<512, 64, 0, stream>>>(slab, obufp);
        gemm_tdm<0,0><<<dim3(1024 / 64, 8192 / 32), 256, 0, stream>>>(
            obufp, 1024, wo_h, 1024, 1024, bo + l * 1024, tmp, nullptr, 1024);
        resid_ln<<<8192, 256, 0, stream>>>(xf, tmp, ln1s + l * 1024, ln1b + l * 1024, xf, xh);
        gemm_tdm<1,1><<<dim3(4096 / 64, 8192 / 32), 256, 0, stream>>>(
            xh, 1024, wff1_h, 1024, 1024, bff1 + l * 4096, nullptr, slab, 4096);
        gemm_tdm<0,0><<<dim3(1024 / 64, 8192 / 32), 256, 0, stream>>>(
            slab, 4096, wff2_h, 4096, 4096, bff2 + l * 1024, tmp, nullptr, 1024);
        resid_ln<<<8192, 256, 0, stream>>>(xf, tmp, ln2s + l * 1024, ln2b + l * 1024, xf, xh);
    }

    extract_h<<<128, 256, 0, stream>>>(xf, h_h);
    gemm_tdm<0,0><<<dim3(128 / 64, 32 / 32), 256, 0, stream>>>(
        h_h, 1024, wproj_h, 1024, 1024, bproj, zp, nullptr, 128);
    sampler<<<8, 256, 0, stream>>>(zp, eps, out_mean, out_logv, out_predz, predzh, z0f);
    times_copy<<<9, 256, 0, stream>>>(ftime, out_times);

    ode_rk4<<<1, 256, 0, stream>>>(ow1h, ob1, owmh, obm, owfh, obf,
                                   z0f, ftime, out_predz, predzh);

    // decoder MLP over pred_z (2080 x 64)
    gemm_tdm<1,1><<<dim3(512 / 64, 2080 / 32), 256, 0, stream>>>(
        predzh, 64, dw1h, 64, 64, db1, nullptr, du0, 512);
    gemm_tdm<1,1><<<dim3(512 / 64, 2080 / 32), 256, 0, stream>>>(
        du0, 512, dwmh, 512, 512, dbm, nullptr, du1, 512);
    gemm_tdm<1,1><<<dim3(512 / 64, 2080 / 32), 256, 0, stream>>>(
        du1, 512, dwmh + 512 * 512, 512, 512, dbm + 512, nullptr, du0, 512);
    gemm_tdm<0,0><<<dim3(64 / 64, 2080 / 32), 256, 0, stream>>>(
        du0, 512, dwouth, 512, 512, dbout, out_predx, nullptr, 64);
}
